// MACELayer_2061584302410
// MI455X (gfx1250) — matrix operand order repaired and checked by reading, not compiled, not run on hardware
//
#include <hip/hip_runtime.h>

// ---------------------------------------------------------------------------
// MACE-style layer for MI455X (gfx1250, wave32).
//  - radial GEMM (E x 64) @ (64 x 320) on v_wmma_f32_16x16x32_f16 with
//    pre-packed fragments (B packed in global, A packed in LDS), vectorized
//    b128 fragment traffic and b128 D-spills (channel-major w tile)
//  - radial tile DMA'd into LDS with TENSOR_LOAD_TO_LDS (TDM) when available
//  - gather / cross / dot message math + f32 atomic scatter (HBM-bound part)
// ---------------------------------------------------------------------------

#define NN 50000
#define EE 800000
#define FF 64
#define SS 10
#define RR 8
#define EPSC 0.25f

static_assert(EE % 64 == 0, "edge tile assumption");

typedef __attribute__((ext_vector_type(16))) _Float16 v16h;
typedef __attribute__((ext_vector_type(8)))  _Float16 v8h;
typedef __attribute__((ext_vector_type(8)))  float    v8f;

#if __has_builtin(__builtin_amdgcn_tensor_load_to_lds)
#define HAVE_TDM 1
typedef unsigned int tdm_v4u __attribute__((ext_vector_type(4)));
typedef int          tdm_v8i __attribute__((ext_vector_type(8)));
typedef int          tdm_v4i __attribute__((ext_vector_type(4)));
#else
#define HAVE_TDM 0
#endif

// ---------------- Kernel 0: pack Wr2 (64x320 f32) into WMMA B-fragments -----
// Wr2p[((nt*2 + kt)*32 + lane)*16 + j] = Wr2[(kt*32 + kb + (j&7) + 16*(j>=8))*320 + nt*16 + n]
// kb = (lane<16 ? 0 : 8), n = lane&15  (CDNA5 16-bit B layout).
__global__ void k_pack_wr2(const float* __restrict__ Wr2, _Float16* __restrict__ Wr2p) {
    int i = blockIdx.x * blockDim.x + threadIdx.x;
    if (i >= 20 * 2 * 32 * 16) return;
    int j    = i & 15;
    int lane = (i >> 4) & 31;
    int rest = i >> 9;
    int kt   = rest & 1;
    int nt   = rest >> 1;
    int kb   = (lane < 16) ? 0 : 8;
    Wr2p[i] = (_Float16)Wr2[(kt * 32 + kb + (j & 7) + ((j >= 8) ? 16 : 0)) * 320 + nt * 16 + (lane & 15)];
}

// ---------------- Kernel 1: node pre: h0 = x0@U0, h1 = x1@U1 ----------------
__global__ void k_node_pre(const float* __restrict__ nf,
                           const float* __restrict__ U0,
                           const float* __restrict__ U1,
                           float* __restrict__ h0,
                           float* __restrict__ h1) {
    __shared__ float x0s[4][64];
    __shared__ float x1s[4][64][3];
    const int slot = threadIdx.x >> 6;
    const int g    = threadIdx.x & 63;
    const int n    = blockIdx.x * 4 + slot;
    if (n < NN) {
        const float* row = nf + (size_t)n * 256;
        x0s[slot][g]    = row[g];
        x1s[slot][g][0] = row[64 + g * 3 + 0];
        x1s[slot][g][1] = row[64 + g * 3 + 1];
        x1s[slot][g][2] = row[64 + g * 3 + 2];
    }
    __syncthreads();
    if (n >= NN) return;
    float a0 = 0.f, ax = 0.f, ay = 0.f, az = 0.f;
    #pragma unroll 8
    for (int f = 0; f < 64; ++f) {
        float u0 = U0[f * 64 + g];
        float u1 = U1[f * 64 + g];
        a0 += x0s[slot][f] * u0;
        ax += x1s[slot][f][0] * u1;
        ay += x1s[slot][f][1] * u1;
        az += x1s[slot][f][2] * u1;
    }
    h0[(size_t)n * 64 + g] = a0;
    float* hp = h1 + ((size_t)n * 64 + g) * 3;
    hp[0] = ax; hp[1] = ay; hp[2] = az;
}

// ---------------- Kernel 2: edge stage --------------------------------------
// block = 128 threads (4 waves), 64 edges per block.
__global__ void k_edge(const float* __restrict__ vectors,
                       const float* __restrict__ radial,
                       const int*   __restrict__ senders,
                       const int*   __restrict__ receivers,
                       const float* __restrict__ Wr1,
                       const _Float16* __restrict__ Wr2p,
                       const float* __restrict__ h0g,
                       const float* __restrict__ h1g,
                       float* __restrict__ a0g,
                       float* __restrict__ a1g) {
    __shared__ __align__(16) float    re_lds[64][8];             // 2 KB (TDM target)
    __shared__ float                  wr1_lds[512];              // 2 KB
    __shared__ __align__(32) _Float16 act_frag[4 * 2 * 32 * 16]; // 8 KB, A-frag order
    __shared__ __align__(16) _Float16 w_cols[4][320][16];        // 40 KB, channel-major
    __shared__ float                  u_lds[64][3];
    __shared__ int                    snd_lds[64];
    __shared__ int                    rcv_lds[64];

    const int tid = threadIdx.x;
    const int e0  = blockIdx.x * 64;

    // ---- radial tile (64 x 8 f32) -> LDS ----
#if HAVE_TDM
    if (tid < 32) {   // one TDM issue from wave 0 (uniform descriptor in SGPRs)
        unsigned int ldsOff = (unsigned int)(uintptr_t)(&re_lds[0][0]); // addr[31:0] = LDS offset
        unsigned long long ga =
            (unsigned long long)(uintptr_t)(radial + (size_t)e0 * RR);
        tdm_v4u g0;
        tdm_v8i g1 = {};
        tdm_v4i g2 = {};
        tdm_v4i g3 = {};
        g0.x = 1u;                                            // count=1, user mode
        g0.y = ldsOff;                                        // lds_addr
        g0.z = (unsigned int)ga;                              // global_addr[31:0]
        g0.w = ((unsigned int)(ga >> 32) & 0x01FFFFFFu)       // global_addr[56:32]
               | 0x80000000u;                                 // type=2 (image)
        g1[0] = (int)(2u << 16);      // data_size = 4 bytes
        g1[1] = (int)(8u << 16);      // tensor_dim0 = 8 elements
        g1[2] = (int)(64u << 16);     // tensor_dim1 = 64 rows
        g1[3] = (int)(8u << 16);      // tile_dim0 = 8
        g1[4] = 64;                   // tile_dim1 = 64 (tile_dim2 = 0)
        g1[5] = 8;                    // tensor_dim0_stride = 8
#if __has_include(<hip/amd_detail/amd_gfx1250_TDM.h>)
        tdm_v8i g4 = {};
        __builtin_amdgcn_tensor_load_to_lds(g0, g1, g2, g3, g4, 0);
#else
        __builtin_amdgcn_tensor_load_to_lds(g0, g1, g2, g3, 0);
#endif
        __builtin_amdgcn_s_wait_tensorcnt(0);
    }
#else
    for (int i = tid; i < 512; i += 128) {
        int el = i >> 3, r = i & 7;
        ((float*)re_lds)[i] = radial[(size_t)(e0 + el) * RR + r];
    }
#endif

    // ---- Wr1 (8x64) -> LDS ----
    for (int i = tid; i < 512; i += 128) wr1_lds[i] = Wr1[i];

    // ---- unit vectors, sender/receiver ids, prefetch gather rows ----
    if (tid < 64) {
        int eg = e0 + tid;
        float vx = vectors[(size_t)eg * 3 + 0];
        float vy = vectors[(size_t)eg * 3 + 1];
        float vz = vectors[(size_t)eg * 3 + 2];
        float r  = sqrtf(vx * vx + vy * vy + vz * vz);
        float inv = 1.f / fmaxf(r, 1e-9f);
        u_lds[tid][0] = vx * inv;
        u_lds[tid][1] = vy * inv;
        u_lds[tid][2] = vz * inv;
        int s = senders[eg];
        snd_lds[tid] = s;
        rcv_lds[tid] = receivers[eg];
        __builtin_prefetch(&h0g[(size_t)s * 64], 0, 0);       // global_prefetch_b8
        __builtin_prefetch(&h1g[(size_t)s * 192], 0, 0);
    }
    __syncthreads();

    // ---- stage A: act = silu(re @ Wr1) written as whole A-fragment lanes ----
    // fragment-lane p = (wv*2 + kt)*32 + ln; element jj -> matrix (m = ln&15,
    // k = kt*32 + (jj&7) + 16*(jj>=8) + 8*(ln>=16)); edge el = wv*16 + m.
    for (int p = tid; p < 256; p += 128) {
        int ln  = p & 31;
        int ktv = p >> 5;
        int kt  = ktv & 1;
        int wv  = ktv >> 1;
        int el  = wv * 16 + (ln & 15);
        int kof = kt * 32 + ((ln >= 16) ? 8 : 0);
        float r0 = re_lds[el][0], r1 = re_lds[el][1], r2 = re_lds[el][2],
              r3 = re_lds[el][3], r4 = re_lds[el][4], r5 = re_lds[el][5],
              r6 = re_lds[el][6], r7 = re_lds[el][7];
        v16h frag;
        #pragma unroll
        for (int jj = 0; jj < 16; ++jj) {
            int o = kof + (jj & 7) + ((jj >= 8) ? 16 : 0);
            float s = r0 * wr1_lds[o]       + r1 * wr1_lds[64 + o] +
                      r2 * wr1_lds[128 + o] + r3 * wr1_lds[192 + o] +
                      r4 * wr1_lds[256 + o] + r5 * wr1_lds[320 + o] +
                      r6 * wr1_lds[384 + o] + r7 * wr1_lds[448 + o];
            float act = s / (1.f + __expf(-s));       // silu
            frag[jj] = (_Float16)act;
        }
        ((v16h*)act_frag)[p] = frag;                  // ds_store_b128 x2
    }
    __syncthreads();

    // ---- stage B: w = act @ Wr2 via WMMA (full EXEC, vectorized fragments) --
    {
        const int lane   = tid & 31;
        const int waveId = tid >> 5;                 // wave owns 16-edge tile
        const v16h* __restrict__ Ap = (const v16h*)act_frag;
        const v16h* __restrict__ Bp = (const v16h*)Wr2p;
        v16h aF0 = Ap[(waveId * 2 + 0) * 32 + lane]; // ds_load_b128 x2
        v16h aF1 = Ap[(waveId * 2 + 1) * 32 + lane];
        const int ncol  = lane & 15;
        const int rbase = (lane < 16) ? 0 : 8;       // D row base
        for (int nt = 0; nt < 20; ++nt) {
            v16h bF0 = Bp[(nt * 2 + 0) * 32 + lane]; // global_load_b128 x2
            v16h bF1 = Bp[(nt * 2 + 1) * 32 + lane];
            v8f acc = {};
            acc = __builtin_amdgcn_wmma_f32_16x16x32_f16(
                false, aF0, false, bF0, (short)0, acc, false, false);
            acc = __builtin_amdgcn_wmma_f32_16x16x32_f16(
                false, aF1, false, bF1, (short)0, acc, false, false);
            v8h hv;
            #pragma unroll
            for (int v = 0; v < 8; ++v) hv[v] = (_Float16)acc[v];
            *(v8h*)&w_cols[waveId][nt * 16 + ncol][rbase] = hv;  // ds_store_b128
        }
    }
    __syncthreads();

    // ---- stage C: messages + scatter-add ----
    for (int i = tid; i < 64 * 64; i += 128) {
        int el = i >> 6, f = i & 63;
        int wv = el >> 4, row = el & 15;
        int s  = snd_lds[el];
        int rr = rcv_lds[el];
        float ux = u_lds[el][0], uy = u_lds[el][1], uz = u_lds[el][2];
        float h0v = h0g[(size_t)s * 64 + f];
        const float* h1p = h1g + ((size_t)s * 64 + f) * 3;
        float hx = h1p[0], hy = h1p[1], hz = h1p[2];
        float w0 = (float)w_cols[wv][f][row];
        float w1 = (float)w_cols[wv][64 + f][row];
        float w2 = (float)w_cols[wv][128 + f][row];
        float w3 = (float)w_cols[wv][192 + f][row];
        float w4 = (float)w_cols[wv][256 + f][row];
        float dot = hx * ux + hy * uy + hz * uz;
        float cx = hy * uz - hz * uy;
        float cy = hz * ux - hx * uz;
        float cz = hx * uy - hy * ux;
        float m0  = w0 * h0v + w1 * dot;
        float w2h = w2 * h0v;
        float m1x = w2h * ux + w3 * hx + w4 * cx;
        float m1y = w2h * uy + w3 * hy + w4 * cy;
        float m1z = w2h * uz + w3 * hz + w4 * cz;
        atomicAdd(&a0g[(size_t)rr * 64 + f], m0);
        float* a1p = a1g + ((size_t)rr * 64 + f) * 3;
        atomicAdd(&a1p[0], m1x);
        atomicAdd(&a1p[1], m1y);
        atomicAdd(&a1p[2], m1z);
    }
}

// ---------------- Kernel 3: node post ---------------------------------------
__global__ void k_node_post(const float* __restrict__ nf,
                            const int*   __restrict__ specie,
                            const float* __restrict__ a0g,
                            const float* __restrict__ a1g,
                            const float* __restrict__ V0,
                            const float* __restrict__ V1,
                            const float* __restrict__ C,
                            const float* __restrict__ G,
                            const float* __restrict__ P0,
                            const float* __restrict__ P1,
                            const float* __restrict__ Wsk0,
                            const float* __restrict__ Wsk1,
                            const float* __restrict__ Wread,
                            float* __restrict__ dout) {
    __shared__ float x0s[4][64];
    __shared__ float x1s[4][64][3];
    __shared__ float t0[4][64];
    __shared__ float t1[4][64][3];
    __shared__ float red[4][64];

    const int slot = threadIdx.x >> 6;
    const int g    = threadIdx.x & 63;
    const int n    = blockIdx.x * 4 + slot;
    const bool ok  = (n < NN);

    if (ok) {
        const float* row = nf + (size_t)n * 256;
        x0s[slot][g]    = row[g];
        x1s[slot][g][0] = row[64 + g * 3 + 0];
        x1s[slot][g][1] = row[64 + g * 3 + 1];
        x1s[slot][g][2] = row[64 + g * 3 + 2];
        t0[slot][g] = EPSC * a0g[(size_t)n * 64 + g];
        const float* ap = a1g + ((size_t)n * 64 + g) * 3;
        t1[slot][g][0] = EPSC * ap[0];
        t1[slot][g][1] = EPSC * ap[1];
        t1[slot][g][2] = EPSC * ap[2];
    }
    __syncthreads();

    float av0 = 0.f, avx = 0.f, avy = 0.f, avz = 0.f;
    if (ok) {
        #pragma unroll 8
        for (int f = 0; f < 64; ++f) {
            float v0 = V0[f * 64 + g];
            float v1 = V1[f * 64 + g];
            av0 += t0[slot][f] * v0;
            avx += t1[slot][f][0] * v1;
            avy += t1[slot][f][1] * v1;
            avz += t1[slot][f][2] * v1;
        }
    }
    __syncthreads();

    int sp = 0;
    if (ok) {
        sp = specie[n];
        float d11 = avx * avx + avy * avy + avz * avz;
        const float* Cs = C + (size_t)sp * 5 * 64;
        const float* Gs = G + (size_t)sp * 4 * 64;
        float c0 = Cs[g], c1 = Cs[64 + g], c2 = Cs[128 + g],
              c3 = Cs[192 + g], c4 = Cs[256 + g];
        float b0 = c0 * av0 + c1 * av0 * av0 + c2 * d11 +
                   c3 * av0 * av0 * av0 + c4 * av0 * d11;
        float gg = Gs[g] + Gs[64 + g] * av0 + Gs[128 + g] * av0 * av0 +
                   Gs[192 + g] * d11;
        t0[slot][g]    = b0;
        t1[slot][g][0] = gg * avx;
        t1[slot][g][1] = gg * avy;
        t1[slot][g][2] = gg * avz;
    }
    __syncthreads();

    float y0 = 0.f, yx = 0.f, yy = 0.f, yz = 0.f;
    if (ok) {
        const float* wk0 = Wsk0 + (size_t)sp * 64 * 64;
        const float* wk1 = Wsk1 + (size_t)sp * 64 * 64;
        #pragma unroll 4
        for (int f = 0; f < 64; ++f) {
            float p0 = P0[f * 64 + g];
            float p1 = P1[f * 64 + g];
            float k0 = wk0[f * 64 + g];
            float k1 = wk1[f * 64 + g];
            y0 += t0[slot][f] * p0 + x0s[slot][f] * k0;
            yx += t1[slot][f][0] * p1 + x1s[slot][f][0] * k1;
            yy += t1[slot][f][1] * p1 + x1s[slot][f][1] * k1;
            yz += t1[slot][f][2] * p1 + x1s[slot][f][2] * k1;
        }
        float* outF = dout + NN + (size_t)n * 256;
        outF[g]              = y0;
        outF[64 + g * 3 + 0] = yx;
        outF[64 + g * 3 + 1] = yy;
        outF[64 + g * 3 + 2] = yz;
        red[slot][g] = y0 * Wread[g];
    }
    __syncthreads();

    if (ok && g == 0) {
        float s = 0.f;
        #pragma unroll 8
        for (int f = 0; f < 64; ++f) s += red[slot][f];
        dout[n] = s;
    }
}

// ---------------------------------------------------------------------------
extern "C" void kernel_launch(void* const* d_in, const int* in_sizes, int n_in,
                              void* d_out, int out_size, void* d_ws, size_t ws_size,
                              hipStream_t stream) {
    const float* vectors    = (const float*)d_in[0];
    const float* node_feats = (const float*)d_in[1];
    const int*   node_spec  = (const int*)d_in[2];
    const float* radial     = (const float*)d_in[3];
    const int*   senders    = (const int*)d_in[4];
    const int*   receivers  = (const int*)d_in[5];
    const float* Wsk0       = (const float*)d_in[6];
    const float* Wsk1       = (const float*)d_in[7];
    const float* U0         = (const float*)d_in[8];
    const float* U1         = (const float*)d_in[9];
    const float* Wr1        = (const float*)d_in[10];
    const float* Wr2        = (const float*)d_in[11];
    const float* V0         = (const float*)d_in[12];
    const float* V1         = (const float*)d_in[13];
    const float* C          = (const float*)d_in[14];
    const float* G          = (const float*)d_in[15];
    const float* P0         = (const float*)d_in[16];
    const float* P1         = (const float*)d_in[17];
    const float* Wread      = (const float*)d_in[18];

    // workspace: h0 | h1 | a0 | a1 | Wr2p   (~103 MB)
    float* h0 = (float*)d_ws;
    float* h1 = h0 + (size_t)NN * FF;
    float* a0 = h1 + (size_t)NN * FF * 3;
    float* a1 = a0 + (size_t)NN * FF;
    _Float16* Wr2p = (_Float16*)(a1 + (size_t)NN * FF * 3);

    k_pack_wr2<<<(20480 + 255) / 256, 256, 0, stream>>>(Wr2, Wr2p);
    k_node_pre<<<(NN + 3) / 4, 256, 0, stream>>>(node_feats, U0, U1, h0, h1);
    hipMemsetAsync(a0, 0, (size_t)NN * FF * 4 * sizeof(float), stream);
    k_edge<<<EE / 64, 128, 0, stream>>>(vectors, radial, senders, receivers,
                                        Wr1, Wr2p, h0, h1, a0, a1);
    k_node_post<<<(NN + 3) / 4, 256, 0, stream>>>(node_feats, node_spec, a0, a1,
                                                  V0, V1, C, G, P0, P1,
                                                  Wsk0, Wsk1, Wread, (float*)d_out);
}
